// MonetLiteSceneGraph_50543175139703
// MI455X (gfx1250) — matrix operand order repaired and checked by reading, not compiled, not run on hardware
//
#include <hip/hip_runtime.h>

// ---------------- problem constants ----------------
#define BB    128
#define CC    128
#define HHh   16
#define WWw   24
#define HW    384
#define NSLOT 10
#define DD2   512
#define HALFC 64
#define PH    22   // H + 6 (7x7 halo)
#define PW    30   // W + 6
#define P2H   20   // H + 4 (5x5 halo)
#define P2W   28   // W + 4

// ---------------- types & helpers ----------------
typedef __attribute__((ext_vector_type(16))) __bf16       v16bf;
typedef __attribute__((ext_vector_type(8)))  float        v8f;
typedef __attribute__((ext_vector_type(8)))  unsigned int v8u;
typedef int v4i_ __attribute__((vector_size(16)));

#define WMMA_BF16(c, a, b)                                                   \
  (c) = __builtin_amdgcn_wmma_f32_16x16x32_bf16(false, (a), false, (b),      \
                                                (short)0, (c), false, false)

#define AS1 __attribute__((address_space(1)))
#define AS3 __attribute__((address_space(3)))

#if __has_builtin(__builtin_amdgcn_global_load_async_to_lds_b128)
#define HAVE_ASYNC_LDS 1
#else
#define HAVE_ASYNC_LDS 0
#endif

// 16-byte global -> LDS copy; async (ASYNCcnt-tracked, no VGPR round trip) when available.
__device__ __forceinline__ void async_copy16(__bf16* lds_dst, const __bf16* gsrc) {
#if HAVE_ASYNC_LDS
  __builtin_amdgcn_global_load_async_to_lds_b128((AS1 v4i_*)gsrc, (AS3 v4i_*)lds_dst, 0, 0);
#else
  *(uint4*)lds_dst = *(const uint4*)gsrc;
#endif
}
__device__ __forceinline__ void async_wait() {
#if HAVE_ASYNC_LDS
#if __has_builtin(__builtin_amdgcn_s_wait_asynccnt)
  __builtin_amdgcn_s_wait_asynccnt(0);
#else
  asm volatile("s_wait_asynccnt 0" ::: "memory");
#endif
#endif
}

__device__ __forceinline__ __bf16 f2bf(float f) {
  unsigned int u = __builtin_bit_cast(unsigned int, f);
  unsigned int r = (u + 0x7FFFu + ((u >> 16) & 1u)) >> 16;
  unsigned short s = (unsigned short)r;
  return __builtin_bit_cast(__bf16, s);
}
__device__ __forceinline__ float bf2f(__bf16 h) {
  unsigned int u = ((unsigned int)__builtin_bit_cast(unsigned short, h)) << 16;
  return __builtin_bit_cast(float, u);
}

// A fragment (16x32 bf16, MxK) from row-major LDS tile, leading dim = lda (bf16 units).
// ISA layout: lane m (0-15) / m+16 hold row m; VGPR v holds K = half*8 + (v&3)*2 + (v>=4)*16.
__device__ __forceinline__ v16bf load_A_frag(const __bf16* p, int lda) {
  int lane = threadIdx.x & 31;
  int m = lane & 15, half = lane >> 4;
  const __bf16* row = p + m * lda + half * 8;
  v8u r;
#pragma unroll
  for (int v = 0; v < 8; ++v) {
    int kb = ((v & 3) * 2) + ((v >> 2) ? 16 : 0);
    r[v] = *(const unsigned int*)(row + kb);
  }
  return __builtin_bit_cast(v16bf, r);
}

// B fragment (32x16 bf16, KxN) taken from LDS holding W^T as [N][K] row-major (ldk).
// ISA layout: lane n / n+16 hold column n; VGPR v holds K = half*8 + (v&3)*2 + (v>=4)*16.
__device__ __forceinline__ v16bf load_B_frag(const __bf16* p, int ldk) {
  int lane = threadIdx.x & 31;
  int n = lane & 15, koff = (lane >> 4) * 8;
  const __bf16* row = p + n * ldk + koff;
  v8u r;
#pragma unroll
  for (int v = 0; v < 8; ++v) r[v] = *(const unsigned int*)(row + ((v & 3) * 2) + ((v >> 2) ? 16 : 0));
  return __builtin_bit_cast(v16bf, r);
}

// ---------------- utility kernels ----------------
__global__ void k_zero32(unsigned int* p, long long n) {
  long long i = (long long)blockIdx.x * blockDim.x + threadIdx.x;
  long long stride = (long long)gridDim.x * blockDim.x;
  for (; i < n; i += stride) p[i] = 0u;
}

__global__ void k_cvt(const float* __restrict__ src, __bf16* __restrict__ dst, int n) {
  int i = blockIdx.x * 256 + threadIdx.x;
  if (i < n) dst[i] = f2bf(src[i]);
}

// w1 raw [64][129][7][7] -> w1bf[o][tap*128 + c] (feat channels only)
__global__ void k_cvt_w1(const float* __restrict__ w1, __bf16* __restrict__ w1bf) {
  int idx = blockIdx.x * 256 + threadIdx.x;
  if (idx >= 64 * 49 * 128) return;
  int c = idx & 127, tap = (idx >> 7) % 49, o = idx / (49 * 128);
  w1bf[idx] = f2bf(w1[((size_t)o * 129 + c) * 49 + tap]);
}
// w2 raw [64][64][7][7] -> w2bf[o][tap*64 + c]
__global__ void k_cvt_w2(const float* __restrict__ w2, __bf16* __restrict__ w2bf) {
  int idx = blockIdx.x * 256 + threadIdx.x;
  if (idx >= 64 * 49 * 64) return;
  int c = idx & 63, tap = (idx >> 6) % 49, o = idx / (49 * 64);
  w2bf[idx] = f2bf(w2[((size_t)o * 64 + c) * 49 + tap]);
}

// ---------------- K1: feature_net 1x1 conv (GEMM 49152x128x128, WMMA) ----------------
__global__ __launch_bounds__(256) void k_feat(const float* __restrict__ x,
                                              const float* __restrict__ wf,
                                              const float* __restrict__ bf_,
                                              float* __restrict__ feat32,
                                              __bf16* __restrict__ featpad) {
  __shared__ __align__(16) __bf16 Abuf[128 * 128];
  __shared__ __align__(16) __bf16 Bbuf[128 * 128];
  int tid = threadIdx.x;
  int p0 = blockIdx.x * 128;                 // 384 blocks, all pixels same batch
  int b0 = p0 / HW, s0 = p0 - b0 * HW;
  for (int i = tid; i < 128 * 128; i += 256) {
    int pp = i & 127, c = i >> 7;
    Abuf[pp * 128 + c] = f2bf(x[((size_t)b0 * CC + c) * HW + s0 + pp]);
  }
  for (int i = tid; i < 128 * 128; i += 256) Bbuf[i] = f2bf(wf[i]);  // [co][c]
  __syncthreads();
  int wave = tid >> 5;
  int mt = wave;                             // wave owns one M tile, loops N tiles
  int lane = tid & 31, nn = lane & 15, mb = (lane >> 4) * 8;
  for (int nt = 0; nt < 8; ++nt) {
    v8f acc0 = {}, acc1 = {};                // dual accumulators: break WMMA RAW chains
#pragma unroll
    for (int kk = 0; kk < 4; kk += 2) {
      v16bf a0 = load_A_frag(&Abuf[(mt * 16) * 128 + kk * 32], 128);
      v16bf b0 = load_B_frag(&Bbuf[(nt * 16) * 128 + kk * 32], 128);
      v16bf a1 = load_A_frag(&Abuf[(mt * 16) * 128 + (kk + 1) * 32], 128);
      v16bf b1 = load_B_frag(&Bbuf[(nt * 16) * 128 + (kk + 1) * 32], 128);
      WMMA_BF16(acc0, a0, b0);
      WMMA_BF16(acc1, a1, b1);
    }
    v8f acc = acc0 + acc1;
    int co = nt * 16 + nn;
    float bias = bf_[co];
#pragma unroll
    for (int v = 0; v < 8; ++v) {
      int s = s0 + mt * 16 + mb + v;
      float val = acc[v] + bias;
      val = val > 0.f ? val : 0.f;
      feat32[((size_t)b0 * HW + s) * CC + co] = val;
      int h = s / WWw, w = s - h * WWw;
      featpad[(((size_t)b0 * PH + (h + 3)) * PW + (w + 3)) * CC + co] = f2bf(val);
    }
  }
}

// ------- K2: loop-invariant feat part of conv1 (7x7, 128->64), implicit GEMM WMMA -------
__global__ __launch_bounds__(256) void k_conv1(const __bf16* __restrict__ featpad,
                                               const __bf16* __restrict__ w1bf,
                                               float* __restrict__ pre1) {
  __shared__ __align__(16) __bf16 slab[7 * 38 * 128];  // padded row slab
  __shared__ __align__(16) __bf16 Bbuf[64 * 128];
  int bh = blockIdx.x, b = bh >> 4, h = bh & 15;
  int tid = threadIdx.x;
  // zero pad columns (pw 30..37) with DS stores
  for (int i = tid; i < 7 * 8 * 16; i += 256) {
    int c8 = i & 15, rest = i >> 4;
    int pw = 30 + (rest % 8), dy = rest / 8;
    *(uint4*)&slab[(dy * 38 + pw) * 128 + c8 * 8] = make_uint4(0, 0, 0, 0);
  }
  // async-stream real columns directly into LDS
  for (int i = tid; i < 7 * 30 * 16; i += 256) {
    int c8 = i & 15, rest = i >> 4;
    int pw = rest % 30, dy = rest / 30;
    async_copy16(&slab[(dy * 38 + pw) * 128 + c8 * 8],
                 &featpad[(((size_t)b * PH + (h + dy)) * PW + pw) * CC + c8 * 8]);
  }
  async_wait();
  int wave = tid >> 5;
  int mt = wave & 1, nt = wave >> 1;
  int w0 = mt * 16;
  v8f acc0 = {}, acc1 = {};
  for (int tap = 0; tap < 49; ++tap) {
    int dy = tap / 7, dx = tap % 7;
    if (tap < 48) __builtin_prefetch(&w1bf[(size_t)(tid & 63) * 6272 + (tap + 1) * 128], 0, 0);
    __syncthreads();
    for (int i = tid; i < 64 * 16; i += 256) {
      int o = i >> 4, c8 = i & 15;
      async_copy16(&Bbuf[o * 128 + c8 * 8], &w1bf[(size_t)o * 6272 + tap * 128 + c8 * 8]);
    }
    async_wait();
    __syncthreads();
    const __bf16* abase = &slab[(dy * 38 + w0 + dx) * 128];
#pragma unroll
    for (int kk = 0; kk < 4; kk += 2) {
      v16bf a0 = load_A_frag(abase + kk * 32, 128);
      v16bf b0 = load_B_frag(&Bbuf[(nt * 16) * 128 + kk * 32], 128);
      v16bf a1 = load_A_frag(abase + (kk + 1) * 32, 128);
      v16bf b1 = load_B_frag(&Bbuf[(nt * 16) * 128 + (kk + 1) * 32], 128);
      WMMA_BF16(acc0, a0, b0);
      WMMA_BF16(acc1, a1, b1);
    }
  }
  v8f acc = acc0 + acc1;
  int lane = tid & 31, nn = lane & 15, mb = (lane >> 4) * 8;
  int o = nt * 16 + nn;
#pragma unroll
  for (int v = 0; v < 8; ++v) {
    int w = w0 + mb + v;
    if (w < WWw) pre1[(((size_t)b * HHh + h) * WWw + w) * HALFC + o] = acc[v];
  }
}

// ------- K3: loop-invariant feat part of residual conv (5x5, 128->1), VALU -------
__global__ __launch_bounds__(256) void k_resfeat(const __bf16* __restrict__ featpad,
                                                 const float* __restrict__ wres,
                                                 float* __restrict__ resf) {
  __shared__ float wl[128 * 25];
  int tid = threadIdx.x;
  for (int i = tid; i < 128 * 25; i += 256) wl[i] = wres[i];  // [c][tap], c<128
  __syncthreads();
  int p = blockIdx.x * 256 + tid;
  if (p >= BB * HW) return;
  int b = p / HW, s = p % HW, h = s / WWw, w = s % WWw;
  float acc = 0.f;
  for (int dy = 0; dy < 5; ++dy)
    for (int dx = 0; dx < 5; ++dx) {
      const __bf16* fp = &featpad[(((size_t)b * PH + (h + dy + 1)) * PW + (w + dx + 1)) * CC];
      int tap = dy * 5 + dx;
      for (int c = 0; c < 128; ++c) acc += bf2f(fp[c]) * wl[c * 25 + tap];
    }
  resf[p] = acc;
}

// ------- K4a (per iter): h1 = relu(pre1 + conv7x7_1ch(scope) + b1) -> bf16 padded -------
__global__ __launch_bounds__(256) void k_h1(const float* __restrict__ pre1,
                                            const float* __restrict__ scope,
                                            const float* __restrict__ w1full,
                                            const float* __restrict__ b1,
                                            __bf16* __restrict__ h1pad) {
  __shared__ float w1s[64 * 49];
  __shared__ float sc[7 * 30];
  int bh = blockIdx.x, b = bh >> 4, h = bh & 15;
  int tid = threadIdx.x;
  for (int i = tid; i < 64 * 49; i += 256) {
    int o = i / 49, tap = i % 49;
    w1s[i] = w1full[((size_t)o * 129 + 128) * 49 + tap];  // scope channel
  }
  for (int i = tid; i < 7 * 30; i += 256) {
    int dy = i / 30, pw = i % 30;
    int hh = h + dy - 3, ww = pw - 3;
    float v = 0.f;
    if (hh >= 0 && hh < HHh && ww >= 0 && ww < WWw) v = scope[(size_t)b * HW + hh * WWw + ww];
    sc[i] = v;
  }
  __syncthreads();
  for (int i = tid; i < 24 * 64; i += 256) {
    int w = i >> 6, o = i & 63;
    float acc = 0.f;
#pragma unroll
    for (int dy = 0; dy < 7; ++dy)
#pragma unroll
      for (int dx = 0; dx < 7; ++dx) acc += sc[dy * 30 + w + dx] * w1s[o * 49 + dy * 7 + dx];
    float val = pre1[(((size_t)b * HHh + h) * WWw + w) * HALFC + o] + acc + b1[o];
    val = val > 0.f ? val : 0.f;
    h1pad[(((size_t)b * PH + (h + 3)) * PW + (w + 3)) * HALFC + o] = f2bf(val);
  }
}

// ------- K4b (per iter): conv2 (7x7, 64->64) implicit GEMM WMMA -> h2 padded -------
__global__ __launch_bounds__(256) void k_conv2(const __bf16* __restrict__ h1pad,
                                               const __bf16* __restrict__ w2bf,
                                               const float* __restrict__ b2,
                                               __bf16* __restrict__ h2pad) {
  __shared__ __align__(16) __bf16 slab[7 * 38 * 64];
  __shared__ __align__(16) __bf16 Bbuf[64 * 64];
  int bh = blockIdx.x, b = bh >> 4, h = bh & 15;
  int tid = threadIdx.x;
  for (int i = tid; i < 7 * 8 * 8; i += 256) {   // zero pad columns
    int c8 = i & 7, rest = i >> 3;
    int pw = 30 + (rest % 8), dy = rest / 8;
    *(uint4*)&slab[(dy * 38 + pw) * 64 + c8 * 8] = make_uint4(0, 0, 0, 0);
  }
  for (int i = tid; i < 7 * 30 * 8; i += 256) {  // async-stream real columns
    int c8 = i & 7, rest = i >> 3;
    int pw = rest % 30, dy = rest / 30;
    async_copy16(&slab[(dy * 38 + pw) * 64 + c8 * 8],
                 &h1pad[(((size_t)b * PH + (h + dy)) * PW + pw) * HALFC + c8 * 8]);
  }
  async_wait();
  int wave = tid >> 5;
  int mt = wave & 1, nt = wave >> 1;
  int w0 = mt * 16;
  v8f acc0 = {}, acc1 = {};
  for (int tap = 0; tap < 49; ++tap) {
    int dy = tap / 7, dx = tap % 7;
    if (tap < 48) __builtin_prefetch(&w2bf[(size_t)(tid & 63) * 3136 + (tap + 1) * 64], 0, 0);
    __syncthreads();
    for (int i = tid; i < 64 * 8; i += 256) {
      int o = i >> 3, c8 = i & 7;
      async_copy16(&Bbuf[o * 64 + c8 * 8], &w2bf[(size_t)o * 3136 + tap * 64 + c8 * 8]);
    }
    async_wait();
    __syncthreads();
    const __bf16* abase = &slab[(dy * 38 + w0 + dx) * 64];
    v16bf a0 = load_A_frag(abase, 64);
    v16bf b0 = load_B_frag(&Bbuf[(nt * 16) * 64], 64);
    v16bf a1 = load_A_frag(abase + 32, 64);
    v16bf b1 = load_B_frag(&Bbuf[(nt * 16) * 64 + 32], 64);
    WMMA_BF16(acc0, a0, b0);
    WMMA_BF16(acc1, a1, b1);
  }
  v8f acc = acc0 + acc1;
  int lane = tid & 31, nn = lane & 15, mb = (lane >> 4) * 8;
  int o = nt * 16 + nn;
  float bias = b2[o];
#pragma unroll
  for (int v = 0; v < 8; ++v) {
    int w = w0 + mb + v;
    if (w < WWw) {
      float val = acc[v] + bias;
      val = val > 0.f ? val : 0.f;
      h2pad[(((size_t)b * P2H + (h + 2)) * P2W + (w + 2)) * HALFC + o] = f2bf(val);
    }
  }
}

// ------- K4c (per iter): logits = conv4(h2) + b4 + res; scope/log_mask recurrence -------
__global__ __launch_bounds__(256) void k_logits(const __bf16* __restrict__ h2pad,
                                                const float* __restrict__ w4,
                                                const float* __restrict__ b4,
                                                const float* __restrict__ resf,
                                                const float* __restrict__ bres,
                                                const float* __restrict__ wres_full,
                                                const float* __restrict__ scope_in,
                                                float* __restrict__ scope_out,
                                                float* __restrict__ logmasks,
                                                int slot, int last) {
  __shared__ float w4l[64 * 25];
  __shared__ float wrs[25];
  __shared__ float sc[5 * 28];
  int bh = blockIdx.x, b = bh >> 4, h = bh & 15;
  int tid = threadIdx.x;
  for (int i = tid; i < 64 * 25; i += 256) w4l[i] = w4[i];           // [c][tap]
  if (tid < 25) wrs[tid] = wres_full[128 * 25 + tid];                // res scope channel
  for (int i = tid; i < 5 * 28; i += 256) {
    int dy = i / 28, pw = i % 28;
    int hh = h + dy - 2, ww = pw - 2;
    float v = 0.f;
    if (hh >= 0 && hh < HHh && ww >= 0 && ww < WWw) v = scope_in[(size_t)b * HW + hh * WWw + ww];
    sc[i] = v;
  }
  __syncthreads();
  int wave = tid >> 5, lane = tid & 31;
  for (int w = wave; w < WWw; w += 8) {
    float part = 0.f;
    int c0 = lane * 2;
    for (int dy = 0; dy < 5; ++dy)
      for (int dx = 0; dx < 5; ++dx) {
        const __bf16* hp = &h2pad[(((size_t)b * P2H + (h + dy)) * P2W + (w + dx)) * HALFC + c0];
        int tap = dy * 5 + dx;
        part += bf2f(hp[0]) * w4l[c0 * 25 + tap] + bf2f(hp[1]) * w4l[(c0 + 1) * 25 + tap];
      }
    for (int off = 16; off >= 1; off >>= 1) part += __shfl_xor(part, off, 32);
    if (lane == 0) {
      float sconv = 0.f;
      for (int dy = 0; dy < 5; ++dy)
        for (int dx = 0; dx < 5; ++dx) sconv += sc[dy * 28 + w + dx] * wrs[dy * 5 + dx];
      size_t ps = (size_t)b * HW + h * WWw + w;
      float logits = part + b4[0] + resf[ps] + bres[0] + sconv;
      float ls_old = scope_in[ps];
      float l1 = logf(1.f + expf(-fabsf(logits)));
      float lsig_pos = fminf(logits, 0.f) - l1;   // logsigmoid(logits)
      float lsig_neg = fminf(-logits, 0.f) - l1;  // logsigmoid(-logits)
      logmasks[((size_t)b * NSLOT + slot) * HW + h * WWw + w] = ls_old + lsig_pos;
      float ns = ls_old + lsig_neg;
      scope_out[ps] = ns;
      if (last) logmasks[((size_t)b * NSLOT + 9) * HW + h * WWw + w] = ns;
    }
  }
}

// ------- K5: masked spatial pooling + validity mask -------
__global__ __launch_bounds__(128) void k_pool(const float* __restrict__ logmasks,
                                              const float* __restrict__ feat32,
                                              const int* __restrict__ objlen,
                                              float* __restrict__ objval,
                                              __bf16* __restrict__ objvalbf) {
  __shared__ float msk[HW];
  int bn = blockIdx.x, b = bn / NSLOT, n = bn % NSLOT;
  int tid = threadIdx.x;
  for (int s = tid; s < HW; s += 128) msk[s] = expf(logmasks[((size_t)b * NSLOT + n) * HW + s]);
  __syncthreads();
  float acc = 0.f;
  const float* fb = &feat32[(size_t)b * HW * CC];
  for (int s = 0; s < HW; ++s) acc += msk[s] * fb[s * CC + tid];
  float valid = (n < objlen[b]) ? 1.f : 0.f;
  acc *= valid;
  objval[(size_t)bn * CC + tid] = acc;
  objvalbf[(size_t)bn * CC + tid] = f2bf(acc);
}

// ------- generic bf16 WMMA GEMM: out[M][N] = A[M][K] * Wt[N][K]^T + bias (+relu) -------
__global__ __launch_bounds__(256) void k_gemm(const __bf16* __restrict__ A,
                                              const __bf16* __restrict__ Wt,
                                              const float* __restrict__ bias,
                                              float* __restrict__ out,
                                              int M, int Nn, int K, int relu) {
  __shared__ __align__(16) __bf16 Abuf[64 * 32];
  __shared__ __align__(16) __bf16 Bbuf[64 * 32];
  int nblk = Nn >> 6;
  int m0 = (blockIdx.x / nblk) * 64;
  int n0 = (blockIdx.x % nblk) * 64;
  int tid = threadIdx.x;
  int wave = tid >> 5;
  int mt = wave & 3, ntb = (wave >> 2) * 2;
  v8f acc0 = {}, acc1 = {};
  for (int k0 = 0; k0 < K; k0 += 32) {
    __syncthreads();
    int r = tid >> 2, c8 = (tid & 3) * 8;
    async_copy16(&Abuf[r * 32 + c8], &A[(size_t)(m0 + r) * K + k0 + c8]);
    async_copy16(&Bbuf[r * 32 + c8], &Wt[(size_t)(n0 + r) * K + k0 + c8]);
    async_wait();
    __syncthreads();
    v16bf a = load_A_frag(&Abuf[mt * 16 * 32], 32);
    v16bf b0 = load_B_frag(&Bbuf[ntb * 16 * 32], 32);
    v16bf b1v = load_B_frag(&Bbuf[(ntb + 1) * 16 * 32], 32);
    WMMA_BF16(acc0, a, b0);
    WMMA_BF16(acc1, a, b1v);
  }
  int lane = tid & 31, nn = lane & 15, mb = (lane >> 4) * 8;
#pragma unroll
  for (int v = 0; v < 8; ++v) {
    int m = m0 + mt * 16 + mb + v;
    int n1 = n0 + ntb * 16 + nn;
    int n2 = n1 + 16;
    float x0 = acc0[v] + bias[n1];
    float x1 = acc1[v] + bias[n2];
    if (relu) { x0 = fmaxf(x0, 0.f); x1 = fmaxf(x1, 0.f); }
    out[(size_t)m * Nn + n1] = x0;
    out[(size_t)m * Nn + n2] = x1;
  }
}

// ------- K7: row L2 normalize; write f32 reps to d_out and bf16 copy for pair GEMMs -------
__global__ __launch_bounds__(256) void k_norm(const float* __restrict__ ru,
                                              float* __restrict__ reps_out,
                                              __bf16* __restrict__ repsbf) {
  __shared__ float red[8];
  int row = blockIdx.x, tid = threadIdx.x;
  float ss = 0.f;
  for (int d = tid; d < DD2; d += 256) {
    float v = ru[(size_t)row * DD2 + d];
    ss += v * v;
  }
  for (int off = 16; off >= 1; off >>= 1) ss += __shfl_xor(ss, off, 32);
  if ((tid & 31) == 0) red[tid >> 5] = ss;
  __syncthreads();
  float tot = red[0] + red[1] + red[2] + red[3] + red[4] + red[5] + red[6] + red[7];
  float scale = 1.f / fmaxf(sqrtf(tot), 1e-12f);
  for (int d = tid; d < DD2; d += 256) {
    float v = ru[(size_t)row * DD2 + d] * scale;
    reps_out[(size_t)row * DD2 + d] = v;
    repsbf[(size_t)row * DD2 + d] = f2bf(v);
  }
}

// ------- K9: pairwise broadcast sum -------
__global__ void k_pairs(const float* __restrict__ o1, const float* __restrict__ o2,
                        float* __restrict__ out) {
  size_t idx = (size_t)blockIdx.x * 256 + threadIdx.x;
  const size_t total = (size_t)BB * NSLOT * NSLOT * DD2;
  if (idx >= total) return;
  int d = (int)(idx & (DD2 - 1));
  size_t q = idx >> 9;            // (b*10+i)*10 + j
  int j = (int)(q % NSLOT);
  size_t bi = q / NSLOT;          // b*10 + i
  out[idx] = o1[bi * DD2 + d] + o2[(bi / NSLOT * NSLOT + j) * DD2 + d];
}

// ---------------- host launch ----------------
extern "C" void kernel_launch(void* const* d_in, const int* in_sizes, int n_in,
                              void* d_out, int out_size, void* d_ws, size_t ws_size,
                              hipStream_t stream) {
  const float* x      = (const float*)d_in[0];
  const int*   objlen = (const int*)d_in[1];
  const float* feat_w = (const float*)d_in[2];
  const float* feat_b = (const float*)d_in[3];
  const float* res_w  = (const float*)d_in[4];
  const float* res_b  = (const float*)d_in[5];
  const float* w1     = (const float*)d_in[6];
  const float* b1     = (const float*)d_in[7];
  const float* w2     = (const float*)d_in[8];
  const float* b2     = (const float*)d_in[9];
  const float* w4     = (const float*)d_in[10];
  const float* b4     = (const float*)d_in[11];
  const float* ofl_w  = (const float*)d_in[12];
  const float* ofl_b  = (const float*)d_in[13];
  const float* o1_w   = (const float*)d_in[14];
  const float* o1_b   = (const float*)d_in[15];
  const float* o2_w   = (const float*)d_in[16];
  const float* o2_b   = (const float*)d_in[17];

  char* ws = (char*)d_ws;
  size_t off = 0;
  auto alloc = [&](size_t bytes) -> char* {
    char* p = ws + off;
    off += (bytes + 255) & ~(size_t)255;
    return p;
  };
  float*  feat32   = (float*)alloc((size_t)BB * HW * CC * 4);
  __bf16* featpad  = (__bf16*)alloc((size_t)BB * PH * PW * CC * 2);
  float*  pre1     = (float*)alloc((size_t)BB * HW * HALFC * 4);
  float*  resf     = (float*)alloc((size_t)BB * HW * 4);
  __bf16* h1pad    = (__bf16*)alloc((size_t)BB * PH * PW * HALFC * 2);
  __bf16* h2pad    = (__bf16*)alloc((size_t)BB * P2H * P2W * HALFC * 2);
  float*  scopeA   = (float*)alloc((size_t)BB * HW * 4);
  float*  scopeB   = (float*)alloc((size_t)BB * HW * 4);
  float*  logmasks = (float*)alloc((size_t)BB * NSLOT * HW * 4);
  float*  objval   = (float*)alloc((size_t)BB * NSLOT * CC * 4);
  __bf16* objvalbf = (__bf16*)alloc((size_t)BB * NSLOT * CC * 2);
  float*  repsun   = (float*)alloc((size_t)BB * NSLOT * DD2 * 4);
  __bf16* repsbf   = (__bf16*)alloc((size_t)BB * NSLOT * DD2 * 2);
  float*  o1buf    = (float*)alloc((size_t)BB * NSLOT * DD2 * 4);
  float*  o2buf    = (float*)alloc((size_t)BB * NSLOT * DD2 * 4);
  __bf16* w1bf     = (__bf16*)alloc((size_t)HALFC * 49 * CC * 2);
  __bf16* w2bf     = (__bf16*)alloc((size_t)HALFC * 49 * HALFC * 2);
  __bf16* oflbf    = (__bf16*)alloc((size_t)DD2 * CC * 2);
  __bf16* ob1bf    = (__bf16*)alloc((size_t)DD2 * DD2 * 2);
  __bf16* ob2bf    = (__bf16*)alloc((size_t)DD2 * DD2 * 2);

  // zero padded/halo buffers and scope (interiors are overwritten; halos must be 0)
  k_zero32<<<2048, 256, 0, stream>>>((unsigned int*)featpad, (long long)BB * PH * PW * CC / 2);
  k_zero32<<<2048, 256, 0, stream>>>((unsigned int*)h1pad, (long long)BB * PH * PW * HALFC / 2);
  k_zero32<<<1024, 256, 0, stream>>>((unsigned int*)h2pad, (long long)BB * P2H * P2W * HALFC / 2);
  k_zero32<<<64, 256, 0, stream>>>((unsigned int*)scopeA, (long long)BB * HW);
  k_zero32<<<64, 256, 0, stream>>>((unsigned int*)scopeB, (long long)BB * HW);

  // weight layout conversions (f32 -> bf16, tap-major for implicit GEMM)
  k_cvt_w1<<<(64 * 49 * 128 + 255) / 256, 256, 0, stream>>>(w1, w1bf);
  k_cvt_w2<<<(64 * 49 * 64 + 255) / 256, 256, 0, stream>>>(w2, w2bf);
  k_cvt<<<(DD2 * CC + 255) / 256, 256, 0, stream>>>(ofl_w, oflbf, DD2 * CC);
  k_cvt<<<(DD2 * DD2 + 255) / 256, 256, 0, stream>>>(o1_w, ob1bf, DD2 * DD2);
  k_cvt<<<(DD2 * DD2 + 255) / 256, 256, 0, stream>>>(o2_w, ob2bf, DD2 * DD2);

  // loop-invariant heavy work
  k_feat<<<384, 256, 0, stream>>>(x, feat_w, feat_b, feat32, featpad);
  k_conv1<<<2048, 256, 0, stream>>>(featpad, w1bf, pre1);
  k_resfeat<<<192, 256, 0, stream>>>(featpad, res_w, resf);

  // MONet recurrence: only scope-dependent work per iteration
  for (int k = 0; k < NSLOT - 1; ++k) {
    float* sin = (k & 1) ? scopeB : scopeA;
    float* sout = (k & 1) ? scopeA : scopeB;
    k_h1<<<2048, 256, 0, stream>>>(pre1, sin, w1, b1, h1pad);
    k_conv2<<<2048, 256, 0, stream>>>(h1pad, w2bf, b2, h2pad);
    k_logits<<<2048, 256, 0, stream>>>(h2pad, w4, b4, resf, res_b, res_w, sin, sout,
                                       logmasks, k, (k == NSLOT - 2) ? 1 : 0);
  }

  k_pool<<<BB * NSLOT, 128, 0, stream>>>(logmasks, feat32, objlen, objval, objvalbf);
  k_gemm<<<(1280 / 64) * (DD2 / 64), 256, 0, stream>>>(objvalbf, oflbf, ofl_b, repsun,
                                                       1280, DD2, CC, 1);
  k_norm<<<1280, 256, 0, stream>>>(repsun, (float*)d_out, repsbf);
  k_gemm<<<(1280 / 64) * (DD2 / 64), 256, 0, stream>>>(repsbf, ob1bf, o1_b, o1buf,
                                                       1280, DD2, DD2, 0);
  k_gemm<<<(1280 / 64) * (DD2 / 64), 256, 0, stream>>>(repsbf, ob2bf, o2_b, o2buf,
                                                       1280, DD2, DD2, 0);
  k_pairs<<<((BB * NSLOT * NSLOT * DD2) + 255) / 256, 256, 0, stream>>>(
      o1buf, o2buf, (float*)d_out + (size_t)BB * NSLOT * DD2);

  (void)in_sizes; (void)n_in; (void)out_size; (void)ws_size;
}
